// EGNNLayer_59639915872350
// MI455X (gfx1250) — compile-verified, matrix operand order checked by reading, not run on hardware
//
#include <hip/hip_runtime.h>
#include <hip/hip_bf16.h>

// ---------------------------------------------------------------------------
// EGNN layer, fused for gfx1250 (MI455X).
//   B=2, Nq=Nr=1024, F=32, He=Ce=64, Hc=32, Hn=Cn=64
// Key algebra:
//   h[c][q,r]   = relu(preq[c][q] + prer[c][r] + w1d[c]*d2(q,r))     (layer1 separable)
//   c_h         = relu(Wcc @ h + bcc),  Wcc = coord_w1 @ edge_w2     (layer2 composed away)
//   x_res_w     = coord_w2 @ c_h + coord_b2
//   m_node      = edge_w2 @ (sum_r h) + 1024*edge_b2                  (linear in h)
// Per-edge GEMM = v_wmma_f32_16x16x32_f16, N = 16 queries, K = 64 channels.
// hsum accumulated in packed f16 (v_pk_add_f16), flushed to f32 every 8 iters.
// Lane-group combine for x_res_w done via final LDS atomics + closed-form
// cb2*sum(dx) term (no per-iteration ds_bpermute).
// ---------------------------------------------------------------------------

typedef __attribute__((ext_vector_type(16))) _Float16 v16h;
typedef __attribute__((ext_vector_type(8)))  float    v8f;

union V16 { uint4 u4[2]; unsigned int u[8]; _Float16 f[16]; v16h v; };
union V8F { float f[8]; v8f v; };

// WMMA f16 half-layout: within a 32-wide K chunk, half index p=2j+half of lane
// group g holds K = 16*(j>>2) + 8*g + 2*(j&3) + half   (ISA 7.12.2 tables)
__device__ __forceinline__ int kf(int j, int h, int g) {
    return 16 * (j >> 2) + 8 * g + 2 * (j & 3) + h;
}

__device__ __forceinline__ unsigned int pack2f(float a, float b) {
    union { _Float16 h; unsigned short s; } ua, ub;
    ua.h = (_Float16)a; ub.h = (_Float16)b;
    return (unsigned int)ua.s | ((unsigned int)ub.s << 16);
}

// ---------------------------------------------------------------------------
// prep_prer: prer[c][r] = sum_i w1[c][32+i]*h_ref[i][r], packed f16 in WMMA-B
// half order: ws layout [b*1024+r][cc][g][j] (32 u32 per r).
// ---------------------------------------------------------------------------
__global__ void prep_prer(const float* __restrict__ h_ref,
                          const float* __restrict__ ew1,
                          unsigned int* __restrict__ prerPk) {
    const int gid = blockIdx.x * 256 + threadIdx.x;   // 0 .. 2047
    const int b = gid >> 10, r = gid & 1023;
    float hr[32];
#pragma unroll
    for (int c = 0; c < 32; ++c) hr[c] = h_ref[(b * 32 + c) * 1024 + r];
    float pr[64];
#pragma unroll 4
    for (int o = 0; o < 64; ++o) {
        float s = 0.f;
#pragma unroll
        for (int c = 0; c < 32; ++c) s += ew1[o * 65 + 32 + c] * hr[c];
        pr[o] = s;
    }
    unsigned int* outp = prerPk + gid * 32;
#pragma unroll
    for (int cc = 0; cc < 2; ++cc)
#pragma unroll
        for (int g = 0; g < 2; ++g)
#pragma unroll
            for (int j = 0; j < 8; ++j)
                outp[(cc * 2 + g) * 8 + j] =
                    pack2f(pr[32 * cc + kf(j, 0, g)], pr[32 * cc + kf(j, 1, g)]);
}

// ---------------------------------------------------------------------------
// prep_params: composite Wcc = coord_w1 @ edge_w2 packed into WMMA-A layout,
// bcc = coord_w1 @ edge_b2 + coord_b1, w1d (dist column of edge_w1) packed.
// ---------------------------------------------------------------------------
__global__ void prep_params(const float* __restrict__ cw1,
                            const float* __restrict__ ew2,
                            const float* __restrict__ eb2,
                            const float* __restrict__ cb1,
                            const float* __restrict__ ew1,
                            unsigned int* __restrict__ wccA,
                            float* __restrict__ bcc,
                            unsigned int* __restrict__ w1dPk) {
    __shared__ float wccS[32 * 64];
    const int t = threadIdx.x;
    for (int idx = t; idx < 2048; idx += 256) {
        const int m = idx >> 6, c = idx & 63;
        float s = 0.f;
#pragma unroll
        for (int e = 0; e < 64; ++e) s += cw1[m * 64 + e] * ew2[e * 64 + c];
        wccS[idx] = s;
    }
    __syncthreads();
    for (int idx = t; idx < 1024; idx += 256) {
        const int j = idx & 7, lane = (idx >> 3) & 31, cc = (idx >> 8) & 1, mt = idx >> 9;
        const int m = (lane & 15) + 16 * mt, g = lane >> 4;
        wccA[idx] = pack2f(wccS[m * 64 + 32 * cc + kf(j, 0, g)],
                           wccS[m * 64 + 32 * cc + kf(j, 1, g)]);
    }
    if (t < 32) {
        float s = 0.f;
#pragma unroll
        for (int e = 0; e < 64; ++e) s += cw1[t * 64 + e] * eb2[e];
        bcc[t] = s + cb1[t];
        const int cc = t >> 4, g = (t >> 3) & 1, j = t & 7;
        w1dPk[t] = pack2f(ew1[(32 * cc + kf(j, 0, g)) * 65 + 64],
                          ew1[(32 * cc + kf(j, 1, g)) * 65 + 64]);
    }
}

// ---------------------------------------------------------------------------
// Main fused kernel: one block per (b, 16-query tile); 8 waves split r.
// ---------------------------------------------------------------------------
__global__ __launch_bounds__(256) void egnn_main(
    const float* __restrict__ x_query, const float* __restrict__ x_ref,
    const float* __restrict__ h_query,
    const float* __restrict__ ew1, const float* __restrict__ eb1,
    const float* __restrict__ ew2, const float* __restrict__ eb2,
    const float* __restrict__ cw2, const float* __restrict__ cb2,
    const float* __restrict__ nw1, const float* __restrict__ nb1,
    const float* __restrict__ nw2, const float* __restrict__ nb2,
    const unsigned int* __restrict__ prerPk,
    const unsigned int* __restrict__ wccA,
    const float* __restrict__ bcc, const unsigned int* __restrict__ w1dPk,
    float* __restrict__ out) {
    const int b = blockIdx.x >> 6;
    const int q0 = (blockIdx.x & 63) * 16;

    __shared__ float preqS[64][16];
    __shared__ float xqS[3][16];
    __shared__ float xrS[3 * 1024];
    __shared__ float hsumS[64][16];
    __shared__ float xupdS[48];
    __shared__ float mnodeS[64][16];
    __shared__ float h1S[64][16];
    __shared__ float bccS[32];
    __shared__ float wc2S[32];

    const int t = threadIdx.x;
    for (int i = t; i < 3 * 1024; i += 256) xrS[i] = x_ref[b * 3072 + i];
    if (t < 48) ((float*)xqS)[t] = x_query[b * 3072 + (t >> 4) * 1024 + q0 + (t & 15)];
    if (t < 32) { bccS[t] = bcc[t]; wc2S[t] = cw2[t]; }
    for (int i = t; i < 1024; i += 256) {
        const int o = i >> 4, nn = i & 15;
        float s = eb1[o];
#pragma unroll
        for (int c = 0; c < 32; ++c)
            s += ew1[o * 65 + c] * h_query[(b * 32 + c) * 1024 + q0 + nn];
        preqS[o][nn] = s;
        ((float*)hsumS)[i] = 0.f;
    }
    if (t < 48) xupdS[t] = 0.f;
    __syncthreads();

    const int lane = t & 31, wave = t >> 5;
    const int n = lane & 15, g = lane >> 4;

    V16 preqPk[2], w1dR[2], wccAr[2][2];
#pragma unroll
    for (int cc = 0; cc < 2; ++cc) {
#pragma unroll
        for (int j = 0; j < 8; ++j) {
            preqPk[cc].f[2 * j + 0] = (_Float16)preqS[32 * cc + kf(j, 0, g)][n];
            preqPk[cc].f[2 * j + 1] = (_Float16)preqS[32 * cc + kf(j, 1, g)][n];
            w1dR[cc].u[j] = w1dPk[(cc * 2 + g) * 8 + j];
        }
    }
#pragma unroll
    for (int mt = 0; mt < 2; ++mt)
#pragma unroll
        for (int cc = 0; cc < 2; ++cc)
#pragma unroll
            for (int j = 0; j < 8; ++j)
                wccAr[mt][cc].u[j] = wccA[((mt * 2 + cc) * 32 + lane) * 8 + j];
    V8F cInit[2];
    float wc2r[16];
#pragma unroll
    for (int mt = 0; mt < 2; ++mt)
#pragma unroll
        for (int v = 0; v < 8; ++v) {
            cInit[mt].f[v] = bccS[16 * mt + v + 8 * g];
            wc2r[mt * 8 + v] = wc2S[16 * mt + v + 8 * g];
        }
    const float xq0 = xqS[0][n], xq1 = xqS[1][n], xq2 = xqS[2][n];
    const float cb2v = cb2[0];
    float hsumA[32];                 // f32 master accumulator
#pragma unroll
    for (int i = 0; i < 32; ++i) hsumA[i] = 0.f;
    V16 hsumH[2];                    // packed-f16 staging accumulator
#pragma unroll
    for (int cc = 0; cc < 2; ++cc)
#pragma unroll
        for (int p = 0; p < 16; ++p) hsumH[cc].f[p] = (_Float16)0.f;
    float xa0 = 0.f, xa1 = 0.f, xa2 = 0.f;     // sum_r part*dx (own lane group)
    float sxr0 = 0.f, sxr1 = 0.f, sxr2 = 0.f;  // sum_r xr (for cb2*sum(dx))

    const unsigned int* prBase = prerPk + (b * 1024) * 32;

    for (int ro = 0; ro < 16; ++ro) {
        for (int ri = 0; ri < 8; ++ri) {           // r = wave + 8*(8*ro+ri)
            const int r = wave + ((ro * 8 + ri) << 3);
            const unsigned int* pr = prBase + r * 32;
            V16 prPk[2];
            prPk[0].u4[0] = *(const uint4*)(pr + g * 8);
            prPk[0].u4[1] = *(const uint4*)(pr + g * 8 + 4);
            prPk[1].u4[0] = *(const uint4*)(pr + 16 + g * 8);
            prPk[1].u4[1] = *(const uint4*)(pr + 16 + g * 8 + 4);
            __builtin_prefetch(pr + 256, 0, 0);

            const float xr0 = xrS[r], xr1 = xrS[1024 + r], xr2 = xrS[2048 + r];
            const float dx0 = xq0 - xr0, dx1 = xq1 - xr1, dx2 = xq2 - xr2;
            sxr0 += xr0; sxr1 += xr1; sxr2 += xr2;
            const float d2 = dx0 * dx0 + dx1 * dx1 + dx2 * dx2;
            const _Float16 d2h = (_Float16)d2;

            V16 hB[2];
#pragma unroll
            for (int cc = 0; cc < 2; ++cc) {
                V16 tmp;
                tmp.v = preqPk[cc].v + prPk[cc].v + w1dR[cc].v * d2h;
#pragma unroll
                for (int p = 0; p < 16; ++p) {                     // ReLU (v_pk_max)
                    _Float16 x = tmp.f[p];
                    tmp.f[p] = (x > (_Float16)0.f) ? x : (_Float16)0.f;
                }
                hB[cc] = tmp;
                hsumH[cc].v += tmp.v;                              // v_pk_add_f16
            }
            // composed coord layer1: c_h[32][16q] = Wcc(32x64) @ h(64x16q) + bcc
            V8F acc[2];
            acc[0] = cInit[0]; acc[1] = cInit[1];
#pragma unroll
            for (int mt = 0; mt < 2; ++mt)
#pragma unroll
                for (int cc = 0; cc < 2; ++cc)
                    acc[mt].v = __builtin_amdgcn_wmma_f32_16x16x32_f16(
                        false, wccAr[mt][cc].v, false, hB[cc].v,
                        (short)0, acc[mt].v, false, false);
            // ReLU + dot with coord_w2 (this lane group's 16 of 32 channels);
            // 4 partials to break the FMA dependency chain.
            float p0 = 0.f, p1 = 0.f, p2 = 0.f, p3 = 0.f;
#pragma unroll
            for (int mt = 0; mt < 2; ++mt) {
#pragma unroll
                for (int v = 0; v < 8; v += 4) {
                    float c0 = acc[mt].f[v + 0]; c0 = c0 > 0.f ? c0 : 0.f;
                    float c1 = acc[mt].f[v + 1]; c1 = c1 > 0.f ? c1 : 0.f;
                    float c2 = acc[mt].f[v + 2]; c2 = c2 > 0.f ? c2 : 0.f;
                    float c3 = acc[mt].f[v + 3]; c3 = c3 > 0.f ? c3 : 0.f;
                    p0 += wc2r[mt * 8 + v + 0] * c0;
                    p1 += wc2r[mt * 8 + v + 1] * c1;
                    p2 += wc2r[mt * 8 + v + 2] * c2;
                    p3 += wc2r[mt * 8 + v + 3] * c3;
                }
            }
            const float part = (p0 + p1) + (p2 + p3);
            xa0 += part * dx0; xa1 += part * dx1; xa2 += part * dx2;
        }
        // flush packed-f16 partial sums (<=8 terms) into f32
#pragma unroll
        for (int cc = 0; cc < 2; ++cc) {
#pragma unroll
            for (int p = 0; p < 16; ++p) {
                hsumA[cc * 16 + p] += (float)hsumH[cc].f[p];
                hsumH[cc].f[p] = (_Float16)0.f;
            }
        }
    }

    // closed-form cb2 * sum_r dx term (once per wave; lane-group 0 only)
    if (g == 0) {
        xa0 += cb2v * (128.f * xq0 - sxr0);
        xa1 += cb2v * (128.f * xq1 - sxr1);
        xa2 += cb2v * (128.f * xq2 - sxr2);
    }

    // reduce per-lane accumulators across waves/lane-groups via LDS
#pragma unroll
    for (int cc = 0; cc < 2; ++cc)
#pragma unroll
        for (int p = 0; p < 16; ++p)
            atomicAdd(&hsumS[32 * cc + kf(p >> 1, p & 1, g)][n], hsumA[cc * 16 + p]);
    atomicAdd(&xupdS[0 * 16 + n], xa0);
    atomicAdd(&xupdS[1 * 16 + n], xa1);
    atomicAdd(&xupdS[2 * 16 + n], xa2);
    __syncthreads();

    if (t < 48) {
        const int d = t >> 4, nn = t & 15;
        out[(b * 3 + d) * 1024 + q0 + nn] = xqS[d][nn] + xupdS[t];
    }
    // m_node = edge_w2 @ hsum + 1024*edge_b2
    for (int i = t; i < 1024; i += 256) {
        const int o = i >> 4, nn = i & 15;
        float s = 1024.f * eb2[o];
#pragma unroll
        for (int c = 0; c < 64; ++c) s += ew2[o * 64 + c] * hsumS[c][nn];
        mnodeS[o][nn] = s;
    }
    __syncthreads();
    // node MLP layer1 (ReLU)
    for (int i = t; i < 1024; i += 256) {
        const int o = i >> 4, nn = i & 15;
        float s = nb1[o];
#pragma unroll
        for (int c = 0; c < 32; ++c)
            s += nw1[o * 96 + c] * h_query[(b * 32 + c) * 1024 + q0 + nn];
#pragma unroll
        for (int c = 0; c < 64; ++c) s += nw1[o * 96 + 32 + c] * mnodeS[c][nn];
        h1S[o][nn] = s > 0.f ? s : 0.f;
    }
    __syncthreads();
    // node MLP layer2 -> h_update
    for (int i = t; i < 1024; i += 256) {
        const int o = i >> 4, nn = i & 15;
        float s = nb2[o];
#pragma unroll
        for (int c = 0; c < 64; ++c) s += nw2[o * 64 + c] * h1S[c][nn];
        out[6144 + (b * 64 + o) * 1024 + q0 + nn] = s;
    }
}

extern "C" void kernel_launch(void* const* d_in, const int* in_sizes, int n_in,
                              void* d_out, int out_size, void* d_ws, size_t ws_size,
                              hipStream_t stream) {
    (void)in_sizes; (void)n_in; (void)out_size; (void)ws_size;
    const float* x_query = (const float*)d_in[0];
    const float* x_ref   = (const float*)d_in[1];
    const float* h_query = (const float*)d_in[2];
    const float* h_ref   = (const float*)d_in[3];
    const float* ew1 = (const float*)d_in[4];
    const float* eb1 = (const float*)d_in[5];
    const float* ew2 = (const float*)d_in[6];
    const float* eb2 = (const float*)d_in[7];
    const float* cw1 = (const float*)d_in[8];
    const float* cb1 = (const float*)d_in[9];
    const float* cw2 = (const float*)d_in[10];
    const float* cb2 = (const float*)d_in[11];
    const float* nw1 = (const float*)d_in[12];
    const float* nb1 = (const float*)d_in[13];
    const float* nw2 = (const float*)d_in[14];
    const float* nb2 = (const float*)d_in[15];
    float* out = (float*)d_out;

    unsigned char* ws = (unsigned char*)d_ws;
    unsigned int* prerPk = (unsigned int*)(ws);                    // 256 KB
    unsigned int* wccA   = (unsigned int*)(ws + 262144);           // 4 KB
    float*        bcc    = (float*)(ws + 262144 + 4096);           // 128 B
    unsigned int* w1dPk  = (unsigned int*)(ws + 262144 + 4096 + 256);

    prep_prer<<<8, 256, 0, stream>>>(h_ref, ew1, prerPk);
    prep_params<<<1, 256, 0, stream>>>(cw1, ew2, eb2, cb1, ew1, wccA, bcc, w1dPk);
    egnn_main<<<128, 256, 0, stream>>>(x_query, x_ref, h_query,
                                       ew1, eb1, ew2, eb2, cw2, cb2,
                                       nw1, nb1, nw2, nb2,
                                       prerPk, wccA, bcc, w1dPk, out);
}
